// MHA_10385230922033
// MI455X (gfx1250) — matrix operand order repaired and checked by reading, not compiled, not run on hardware
//
#include <hip/hip_runtime.h>
#include <hip/hip_bf16.h>

typedef __attribute__((ext_vector_type(16))) _Float16 v16h;
typedef __attribute__((ext_vector_type(8)))  _Float16 v8h;
typedef __attribute__((ext_vector_type(8)))  float    v8f;

#define TT 4096
#define DMODEL 1024
#define NHEAD 8
#define HDIM 64
#define NKV 512   // NHEAD*HDIM

// LDS row pads (halves) chosen so lane->bank = n*20 mod 64 / n*36 mod 64 are all distinct
#define LDP 40    // for 32-half rows (GEMM tiles, V tile)
#define LDK 72    // for 64-half rows (K tile)

// ---------------- CDNA5 async global->LDS copy (ASYNCcnt path) ----------------

__device__ __forceinline__ void async_cp16(const _Float16* g, _Float16* l) {
    unsigned l32 = (unsigned)(size_t)l;   // generic LDS ptr: low 32 bits are the LDS byte offset
    asm volatile("global_load_async_to_lds_b128 %0, %1, off"
                 :: "v"(l32), "v"(g) : "memory");
}
__device__ __forceinline__ void wait_async() {
    asm volatile("s_wait_asynccnt 0" ::: "memory");
}

// ---------------- WMMA fragment helpers (gfx1250 wave32 layouts) ----------------

__device__ __forceinline__ v16h cat16(v8h lo, v8h hi) {
    v16h r;
#pragma unroll
    for (int i = 0; i < 8; ++i) { r[i] = lo[i]; r[i + 8] = hi[i]; }
    return r;
}

// A matrix 16x32 f16, row-major, leading dim `ld` (halves).
__device__ __forceinline__ v16h load_a_frag(const _Float16* base, int ld) {
    int lane = threadIdx.x & 31;
    int m  = lane & 15;
    int kb = (lane >> 4) * 8;
    const _Float16* p = base + m * ld + kb;
    v8h lo = *(const v8h*)(p);
    v8h hi = *(const v8h*)(p + 16);
    return cat16(lo, hi);
}

// B matrix 32x16 f16 supplied TRANSPOSED: Bt[n][k], leading dim `ld` (halves).
__device__ __forceinline__ v16h load_b_frag(const _Float16* baseT, int ld) {
    int lane = threadIdx.x & 31;
    int n  = lane & 15;
    int kb = (lane >> 4) * 8;
    const _Float16* p = baseT + n * ld + kb;
    v8h lo = *(const v8h*)(p);
    v8h hi = *(const v8h*)(p + 16);
    return cat16(lo, hi);
}

__device__ __forceinline__ v8f wmma_f16(v16h a, v16h b, v8f c) {
    return __builtin_amdgcn_wmma_f32_16x16x32_f16(false, a, false, b, (short)0, c, false, false);
}

// ---------------- Conversion / packing kernels ----------------

__global__ void __launch_bounds__(256)
cvt_f32_to_f16(const float* __restrict__ in, _Float16* __restrict__ out, int n) {
    int i = blockIdx.x * 256 + threadIdx.x;
    if (i < n) out[i] = (_Float16)in[i];
}

// in: [K,N] f32 row-major -> outT: [N,K] f16 row-major
__global__ void __launch_bounds__(256)
transpose_cvt(const float* __restrict__ in, _Float16* __restrict__ outT, int K, int N) {
    int i = blockIdx.x * 256 + threadIdx.x;
    if (i < K * N) {
        int k = i / N, n = i - k * N;
        outT[n * K + k] = (_Float16)in[i];
    }
}

// Reference's unusual reshape: Xh[h,t,d] = Xs[h*512 + d*8 + t/512, t%512]
// Also builds V transposed per head: Vt[h][d][t].
__global__ void __launch_bounds__(256)
gather_heads(const _Float16* __restrict__ Qs, const _Float16* __restrict__ Ks,
             const _Float16* __restrict__ Vs,
             _Float16* __restrict__ Qh, _Float16* __restrict__ Kh,
             _Float16* __restrict__ Vt) {
    int i = blockIdx.x * 256 + threadIdx.x;          // over NHEAD*TT*HDIM
    if (i >= NHEAD * TT * HDIM) return;
    int d = i & (HDIM - 1);
    int t = (i >> 6) & (TT - 1);
    int h = i >> 18;
    int src = (h * 512 + d * 8 + (t >> 9)) * 512 + (t & 511);
    Qh[i] = Qs[src];
    Kh[i] = Ks[src];
    Vt[((h * HDIM + d) << 12) + t] = Vs[src];
}

// ---------------- WMMA GEMM: C[M,N] = A[M,K] @ Bt[N,K]^T + bias ----------------
// 128 threads (4 waves), 64x64 tile; double-buffered async LDS staging of A/B K-slices.

template <bool F16OUT>
__global__ void __launch_bounds__(128)
gemm_wmma(const _Float16* __restrict__ A, const _Float16* __restrict__ Bt,
          const float* __restrict__ bias,
          _Float16* __restrict__ C16, float* __restrict__ C32,
          int M, int N, int K) {
    __shared__ __align__(16) _Float16 As[2][64 * LDP];
    __shared__ __align__(16) _Float16 Bs[2][64 * LDP];
    int tid  = threadIdx.x;
    int wave = tid >> 5, lane = tid & 31;
    int row0 = blockIdx.x * 64;
    int col0 = blockIdx.y * 64;

    // stage one 64x32 K-slice of A and Bt into LDS buffer `b` via async DMA
    auto stage = [&](int b, int k0) {
#pragma unroll
        for (int it = 0; it < 2; ++it) {
            int i = tid + it * 128;      // i in [0,256)
            int r = i >> 2;
            int c = (i & 3) * 8;
            async_cp16(&A [(size_t)(row0 + r) * K + k0 + c], &As[b][r * LDP + c]);
            async_cp16(&Bt[(size_t)(col0 + r) * K + k0 + c], &Bs[b][r * LDP + c]);
        }
    };

    v8f acc[4] = {};
    int buf = 0;
    stage(0, 0);
    for (int k0 = 0; k0 < K; k0 += 32) {
        wait_async();
        __syncthreads();
        if (k0 + 32 < K) stage(buf ^ 1, k0 + 32);   // prefetch overlaps the WMMAs below
        // batch all fragment loads (one ds clause), then the 4 WMMAs
        v16h a = load_a_frag(&As[buf][(wave * 16) * LDP], LDP);
        v16h b0 = load_b_frag(&Bs[buf][ 0 * LDP], LDP);
        v16h b1 = load_b_frag(&Bs[buf][16 * LDP], LDP);
        v16h b2 = load_b_frag(&Bs[buf][32 * LDP], LDP);
        v16h b3 = load_b_frag(&Bs[buf][48 * LDP], LDP);
        acc[0] = wmma_f16(a, b0, acc[0]);
        acc[1] = wmma_f16(a, b1, acc[1]);
        acc[2] = wmma_f16(a, b2, acc[2]);
        acc[3] = wmma_f16(a, b3, acc[3]);
        __syncthreads();
        buf ^= 1;
    }
    int halfg = lane >> 4;
    int nlane = lane & 15;
#pragma unroll
    for (int nt = 0; nt < 4; ++nt) {
        int col = col0 + nt * 16 + nlane;
        float bv = bias ? bias[col] : 0.0f;
#pragma unroll
        for (int r = 0; r < 8; ++r) {
            int row = row0 + wave * 16 + r + halfg * 8;
            float v = acc[nt][r] + bv;
            if (F16OUT) C16[(size_t)row * N + col] = (_Float16)v;
            else        C32[(size_t)row * N + col] = v;
        }
    }
}

// ---------------- Flash attention per head ----------------
// Grid: (TT/64, NHEAD), 128 threads (4 waves); each wave owns 16 query rows.
// 32-key blocks: K/V tiles async-staged to LDS (double-buffered, shared by all 4 waves),
// S = Q K^T (4 WMMAs), scale+mask, online softmax, P via LDS, O += P V (4 WMMAs).

__global__ void __launch_bounds__(128)
flash_attn(const _Float16* __restrict__ Qh, const _Float16* __restrict__ Kh,
           const _Float16* __restrict__ Vt, const float* __restrict__ mask,
           _Float16* __restrict__ A16) {
    __shared__ __align__(16) _Float16 Ks_lds[2][32 * LDK];   // [key][d]
    __shared__ __align__(16) _Float16 Vs_lds[2][64 * LDP];   // [dv][key]
    __shared__ __align__(16) _Float16 P_lds[4][16 * 32];
    int h    = blockIdx.y;
    int tid  = threadIdx.x;
    int wave = tid >> 5;
    int lane = tid & 31;
    int halfg = lane >> 4;
    int nlane = lane & 15;
    int q0 = blockIdx.x * 64 + wave * 16;

    const _Float16* Qhead = Qh + (size_t)h * TT * HDIM;
    const _Float16* Khead = Kh + (size_t)h * TT * HDIM;
    const _Float16* Vhead = Vt + (size_t)h * HDIM * TT;
    _Float16* P = &P_lds[wave][0];

    auto stage = [&](int b, int kb) {
#pragma unroll
        for (int it = 0; it < 2; ++it) {
            int i = tid + it * 128;              // i in [0,256)
            int kr = i >> 3, kc = (i & 7) * 8;   // K tile: 32 rows x 64 halves
            async_cp16(Khead + (size_t)(kb + kr) * HDIM + kc, &Ks_lds[b][kr * LDK + kc]);
            int vr = i >> 2, vc = (i & 3) * 8;   // V tile: 64 rows x 32 halves
            async_cp16(Vhead + (size_t)vr * TT + kb + vc, &Vs_lds[b][vr * LDP + vc]);
        }
    };

    v16h qa0 = load_a_frag(Qhead + (size_t)q0 * HDIM,      HDIM);  // d 0..31
    v16h qa1 = load_a_frag(Qhead + (size_t)q0 * HDIM + 32, HDIM);  // d 32..63

    v8f o[4] = {};
    float rmax[8], rsum[8];
#pragma unroll
    for (int r = 0; r < 8; ++r) { rmax[r] = -3.0e38f; rsum[r] = 0.0f; }
    const float scale = 0.125f;  // 1/sqrt(64)

    int buf = 0;
    stage(0, 0);
    for (int kb = 0; kb < TT; kb += 32) {
        wait_async();
        __syncthreads();
        if (kb + 32 < TT) stage(buf ^ 1, kb + 32);   // async DMA overlaps compute below
        // ---- scores: batch all 4 K-fragment loads, then 4 WMMAs ----
        v16h kb00 = load_b_frag(&Ks_lds[buf][0],             LDK);
        v16h kb01 = load_b_frag(&Ks_lds[buf][32],            LDK);
        v16h kb10 = load_b_frag(&Ks_lds[buf][16 * LDK],      LDK);
        v16h kb11 = load_b_frag(&Ks_lds[buf][16 * LDK + 32], LDK);
        v8f s0 = {}, s1 = {};
        s0 = wmma_f16(qa0, kb00, s0);
        s1 = wmma_f16(qa0, kb10, s1);
        s0 = wmma_f16(qa1, kb01, s0);
        s1 = wmma_f16(qa1, kb11, s1);
        // ---- scale + mask + online softmax ----
#pragma unroll
        for (int r = 0; r < 8; ++r) {
            int row = q0 + r + halfg * 8;
            s0[r] = s0[r] * scale + mask[(size_t)row * TT + kb + nlane];
            s1[r] = s1[r] * scale + mask[(size_t)row * TT + kb + 16 + nlane];
        }
#pragma unroll
        for (int r = 0; r < 8; ++r) {
            float m = fmaxf(s0[r], s1[r]);
            m = fmaxf(m, __shfl_xor(m, 1));
            m = fmaxf(m, __shfl_xor(m, 2));
            m = fmaxf(m, __shfl_xor(m, 4));
            m = fmaxf(m, __shfl_xor(m, 8));
            float newm = fmaxf(rmax[r], m);
            float corr = __expf(rmax[r] - newm);
            rmax[r] = newm;
            rsum[r] *= corr;
            o[0][r] *= corr; o[1][r] *= corr; o[2][r] *= corr; o[3][r] *= corr;
            float p0 = __expf(s0[r] - newm);
            float p1 = __expf(s1[r] - newm);
            float ps = p0 + p1;
            ps += __shfl_xor(ps, 1);
            ps += __shfl_xor(ps, 2);
            ps += __shfl_xor(ps, 4);
            ps += __shfl_xor(ps, 8);
            rsum[r] += ps;
            int prow = r + halfg * 8;
            P[prow * 32 + nlane]      = (_Float16)p0;
            P[prow * 32 + 16 + nlane] = (_Float16)p1;
        }
        // same-wave LDS RAW on P: DS ops in-order, make the dependency explicit
        asm volatile("s_wait_dscnt 0" ::: "memory");
        // ---- O += P @ V: batch P + all 4 V-fragment loads, then 4 WMMAs ----
        v16h pa  = load_a_frag(P, 32);
        v16h vb0 = load_b_frag(&Vs_lds[buf][ 0 * LDP], LDP);
        v16h vb1 = load_b_frag(&Vs_lds[buf][16 * LDP], LDP);
        v16h vb2 = load_b_frag(&Vs_lds[buf][32 * LDP], LDP);
        v16h vb3 = load_b_frag(&Vs_lds[buf][48 * LDP], LDP);
        o[0] = wmma_f16(pa, vb0, o[0]);
        o[1] = wmma_f16(pa, vb1, o[1]);
        o[2] = wmma_f16(pa, vb2, o[2]);
        o[3] = wmma_f16(pa, vb3, o[3]);
        __syncthreads();
        buf ^= 1;
    }
    // ---- normalize + write A16[t][h*64 + d] ----
#pragma unroll
    for (int nt = 0; nt < 4; ++nt) {
        int col = h * HDIM + nt * 16 + nlane;
#pragma unroll
        for (int r = 0; r < 8; ++r) {
            int row = q0 + r + halfg * 8;
            A16[(size_t)row * NKV + col] = (_Float16)(o[nt][r] / rsum[r]);
        }
    }
}

// ---------------- host orchestration ----------------

extern "C" void kernel_launch(void* const* d_in, const int* in_sizes, int n_in,
                              void* d_out, int out_size, void* d_ws, size_t ws_size,
                              hipStream_t stream) {
    (void)in_sizes; (void)n_in; (void)out_size; (void)ws_size;
    const float* Q    = (const float*)d_in[0];
    const float* K    = (const float*)d_in[1];
    const float* V    = (const float*)d_in[2];
    const float* mask = (const float*)d_in[3];
    const float* Wq   = (const float*)d_in[4];
    const float* bq   = (const float*)d_in[5];
    const float* Wk   = (const float*)d_in[6];
    const float* bk   = (const float*)d_in[7];
    const float* Wv   = (const float*)d_in[8];
    const float* bv   = (const float*)d_in[9];
    const float* Wo   = (const float*)d_in[10];
    const float* bo   = (const float*)d_in[11];
    float* out = (float*)d_out;

    char* w = (char*)d_ws;
    _Float16* q16  = (_Float16*)w; w += (size_t)TT * DMODEL * 2;
    _Float16* k16  = (_Float16*)w; w += (size_t)TT * DMODEL * 2;
    _Float16* v16  = (_Float16*)w; w += (size_t)TT * DMODEL * 2;
    _Float16* wqT  = (_Float16*)w; w += (size_t)DMODEL * NKV * 2;
    _Float16* wkT  = (_Float16*)w; w += (size_t)DMODEL * NKV * 2;
    _Float16* wvT  = (_Float16*)w; w += (size_t)DMODEL * NKV * 2;
    _Float16* woT  = (_Float16*)w; w += (size_t)NKV * DMODEL * 2;
    _Float16* qs16 = (_Float16*)w; w += (size_t)TT * NKV * 2;
    _Float16* ks16 = (_Float16*)w; w += (size_t)TT * NKV * 2;
    _Float16* vs16 = (_Float16*)w; w += (size_t)TT * NKV * 2;
    _Float16* qh   = (_Float16*)w; w += (size_t)NHEAD * TT * HDIM * 2;
    _Float16* kh   = (_Float16*)w; w += (size_t)NHEAD * TT * HDIM * 2;
    _Float16* vt   = (_Float16*)w; w += (size_t)NHEAD * TT * HDIM * 2;
    _Float16* a16  = (_Float16*)w; w += (size_t)TT * NKV * 2;

    // 1) convert activations
    int nact = TT * DMODEL;
    cvt_f32_to_f16<<<(nact + 255) / 256, 256, 0, stream>>>(Q, q16, nact);
    cvt_f32_to_f16<<<(nact + 255) / 256, 256, 0, stream>>>(K, k16, nact);
    cvt_f32_to_f16<<<(nact + 255) / 256, 256, 0, stream>>>(V, v16, nact);
    // 2) transpose+convert weights
    int nw = DMODEL * NKV;
    transpose_cvt<<<(nw + 255) / 256, 256, 0, stream>>>(Wq, wqT, DMODEL, NKV);
    transpose_cvt<<<(nw + 255) / 256, 256, 0, stream>>>(Wk, wkT, DMODEL, NKV);
    transpose_cvt<<<(nw + 255) / 256, 256, 0, stream>>>(Wv, wvT, DMODEL, NKV);
    transpose_cvt<<<(nw + 255) / 256, 256, 0, stream>>>(Wo, woT, NKV, DMODEL);
    // 3) projections
    dim3 gp(TT / 64, NKV / 64);
    gemm_wmma<true><<<gp, 128, 0, stream>>>(q16, wqT, bq, qs16, nullptr, TT, NKV, DMODEL);
    gemm_wmma<true><<<gp, 128, 0, stream>>>(k16, wkT, bk, ks16, nullptr, TT, NKV, DMODEL);
    gemm_wmma<true><<<gp, 128, 0, stream>>>(v16, wvT, bv, vs16, nullptr, TT, NKV, DMODEL);
    // 4) head gather (reference's unusual reshape) + V transpose
    int ng = NHEAD * TT * HDIM;
    gather_heads<<<(ng + 255) / 256, 256, 0, stream>>>(qs16, ks16, vs16, qh, kh, vt);
    // 5) flash attention
    dim3 ga(TT / 64, NHEAD);
    flash_attn<<<ga, 128, 0, stream>>>(qh, kh, vt, mask, a16);
    // 6) output projection -> f32
    dim3 go(TT / 64, DMODEL / 64);
    gemm_wmma<false><<<go, 128, 0, stream>>>(a16, woT, bo, nullptr, out, TT, DMODEL, NKV);
}
